// GraphSAGE_56659208568913
// MI455X (gfx1250) — compile-verified, matrix operand order checked by reading, not run on hardware
//
#include <hip/hip_runtime.h>
#include <math.h>

#define NN 50000
#define KDEG 16
#define CD 128
#define NT 3125          // NN / 16
#define JKH 192

typedef __attribute__((ext_vector_type(16))) __bf16 v16bf;
typedef __attribute__((ext_vector_type(16))) unsigned short v16us;
typedef __attribute__((ext_vector_type(8)))  float v8f;
typedef __attribute__((ext_vector_type(4)))  unsigned int v4u;
typedef __attribute__((ext_vector_type(8)))  int v8i;
typedef __attribute__((ext_vector_type(4)))  int v4i;

union U8 { uint4 v; unsigned short s[8]; };

__device__ __forceinline__ unsigned short f2bf(float f) {
  union { float f; unsigned u; } v; v.f = f;
  unsigned r = v.u + 0x7FFFu + ((v.u >> 16) & 1u);
  return (unsigned short)(r >> 16);
}
__device__ __forceinline__ float bf2f(unsigned short h) {
  union { unsigned u; float f; } v; v.u = ((unsigned)h) << 16;
  return v.f;
}
__device__ __forceinline__ float sigm(float x) { return 1.0f / (1.0f + expf(-x)); }

// A-fragment: rows = (row0 + lane&15); element e -> K = kb + (lane&16?8:0) + (e<8 ? e : e+8)
// B-fragment is the same pattern with rows = output-neuron n, since B(k,n) = W[n][k] (row-major W).
__device__ __forceinline__ v16bf load_frag(const unsigned short* base, int ld, int row0, int kb, int lane) {
  const unsigned short* p = base + (size_t)(row0 + (lane & 15)) * ld + kb + ((lane & 16) ? 8 : 0);
  v16us t;
#pragma unroll
  for (int e = 0; e < 8; ++e) { t[e] = p[e]; t[e + 8] = p[e + 16]; }
  return __builtin_bit_cast(v16bf, t);
}

__device__ __forceinline__ v8f wmma_bf16(v16bf a, v16bf b, v8f c) {
  return __builtin_amdgcn_wmma_f32_16x16x32_bf16(false, a, false, b, (short)0, c, false, false);
}

// ---------------- TDM: 2D tile (rows x cols of 2-byte elems) global -> LDS, with
// 16B padding inserted after every 256B row so LDS row stride is (cols+8) elems. ----------------
#if __has_builtin(__builtin_amdgcn_tensor_load_to_lds)
#define HAVE_TDM 1
#if defined(__clang_major__) && (__clang_major__ >= 23)
#define TDM_CALL(g0, g1, g2, g3) \
  __builtin_amdgcn_tensor_load_to_lds(g0, g1, g2, g3, (v8i){0, 0, 0, 0, 0, 0, 0, 0}, 0)
#else
#define TDM_CALL(g0, g1, g2, g3) __builtin_amdgcn_tensor_load_to_lds(g0, g1, g2, g3, 0)
#endif
__device__ __forceinline__ void tdm_load_2d_pad(const unsigned short* gsrc,
                                                unsigned int lds_byte_addr,
                                                int rows, int cols) {
  unsigned long long ga = (unsigned long long)(size_t)gsrc;
  v4u g0;
  g0[0] = 1u;                                                // count=1 (valid), user mode
  g0[1] = lds_byte_addr;                                     // lds_addr
  g0[2] = (unsigned)(ga & 0xFFFFFFFFu);                      // global_addr[31:0]
  g0[3] = (unsigned)((ga >> 32) & 0x1FFFFFFu) | (2u << 30);  // global_addr[56:32], type=2
  v8i g1;
  g1[0] = (1 << 16) | (1 << 20) | (5 << 22) | (3 << 25);     // data_size=2B, pad_en, ivl=64DW, amt=4DW
  g1[1] = (cols & 0xFFFF) << 16;                             // tensor_dim0[15:0] @ bits 63:48
  g1[2] = ((cols >> 16) & 0xFFFF) | ((rows & 0xFFFF) << 16); // dim0 hi | tensor_dim1 lo
  g1[3] = ((rows >> 16) & 0xFFFF) | ((cols & 0xFFFF) << 16); // dim1 hi | tile_dim0
  g1[4] = (rows & 0xFFFF);                                   // tile_dim1 (tile_dim2=0)
  g1[5] = cols;                                              // tensor_dim0_stride[31:0]
  g1[6] = 0;
  g1[7] = 0;                                                 // dim1_stride unused (2D tile)
  v4i z4 = {0, 0, 0, 0};
  TDM_CALL(g0, g1, z4, z4);
}
#else
#define HAVE_TDM 0
#endif

// ---------------- conversion ----------------
__global__ void k_cvt(const float* __restrict__ s, unsigned short* __restrict__ d, int n) {
  int i = blockIdx.x * blockDim.x + threadIdx.x;
  if (i < n) d[i] = f2bf(s[i]);
}

// ---------------- aggregations (16 nodes/block, b128 row gathers) ----------------
__global__ __launch_bounds__(256) void k_mean_agg(const unsigned short* __restrict__ xb,
                                                  const long long* __restrict__ src,
                                                  unsigned short* __restrict__ agg) {
  int tid = threadIdx.x;
  int node = blockIdx.x * 16 + (tid >> 4);
  int fb = (tid & 15) * 8;
  const long long* sp = src + (size_t)node * KDEG;
  float acc[8];
#pragma unroll
  for (int e = 0; e < 8; ++e) acc[e] = 0.f;
  for (int t = 0; t < KDEG; ++t) {
    long long s = sp[t];
    if (t + 1 < KDEG) __builtin_prefetch(xb + (size_t)sp[t + 1] * CD + fb, 0, 1);
    U8 u; u.v = *(const uint4*)(xb + (size_t)s * CD + fb);
#pragma unroll
    for (int e = 0; e < 8; ++e) acc[e] += bf2f(u.s[e]);
  }
  U8 o;
#pragma unroll
  for (int e = 0; e < 8; ++e) o.s[e] = f2bf(acc[e] * (1.0f / KDEG));
  *(uint4*)(agg + (size_t)node * CD + fb) = o.v;
}

__global__ __launch_bounds__(256) void k_max_agg(const unsigned short* __restrict__ hb,
                                                 const long long* __restrict__ src,
                                                 unsigned short* __restrict__ agg) {
  int tid = threadIdx.x;
  int node = blockIdx.x * 16 + (tid >> 4);
  int fb = (tid & 15) * 8;
  const long long* sp = src + (size_t)node * KDEG;
  float acc[8];
#pragma unroll
  for (int e = 0; e < 8; ++e) acc[e] = -3.0e38f;
  for (int t = 0; t < KDEG; ++t) {
    long long s = sp[t];
    if (t + 1 < KDEG) __builtin_prefetch(hb + (size_t)sp[t + 1] * CD + fb, 0, 1);
    U8 u; u.v = *(const uint4*)(hb + (size_t)s * CD + fb);
#pragma unroll
    for (int e = 0; e < 8; ++e) acc[e] = fmaxf(acc[e], bf2f(u.s[e]));
  }
  U8 o;
#pragma unroll
  for (int e = 0; e < 8; ++e) o.s[e] = f2bf(acc[e] < -1.0e37f ? 0.f : acc[e]);
  *(uint4*)(agg + (size_t)node * CD + fb) = o.v;
}

// ---------------- fused SAGE combine: act(A1@W1.T + b + A2@W2.T) ----------------
__global__ __launch_bounds__(256) void k_sage_gemm(
    const unsigned short* __restrict__ A1, const unsigned short* __restrict__ W1,
    const float* __restrict__ bias,
    const unsigned short* __restrict__ A2, const unsigned short* __restrict__ W2,
    unsigned short* __restrict__ outb, int relu) {
  __shared__ unsigned short A1t[16 * 136];
  __shared__ unsigned short A2t[16 * 136];
  int node0 = blockIdx.x * 16;
  int tid = threadIdx.x;
  {
    int row = tid >> 4, cb = (tid & 15) * 8;
    *(uint4*)(A1t + row * 136 + cb) = *(const uint4*)(A1 + (size_t)(node0 + row) * CD + cb);
    *(uint4*)(A2t + row * 136 + cb) = *(const uint4*)(A2 + (size_t)(node0 + row) * CD + cb);
  }
  __syncthreads();
  int wv = tid >> 5, lane = tid & 31;
  int nb = wv * 16;
  v8f acc = {};
#pragma unroll
  for (int kb = 0; kb < CD; kb += 32) {
    acc = wmma_bf16(load_frag(A1t, 136, 0, kb, lane), load_frag(W1, CD, nb, kb, lane), acc);
    acc = wmma_bf16(load_frag(A2t, 136, 0, kb, lane), load_frag(W2, CD, nb, kb, lane), acc);
  }
  int j = nb + (lane & 15);
  float bj = bias[j];
  int mo = (lane & 16) ? 8 : 0;
#pragma unroll
  for (int v = 0; v < 8; ++v) {
    float val = acc[v] + bj;
    if (relu) val = fmaxf(val, 0.f);
    outb[(size_t)(node0 + v + mo) * CD + j] = f2bf(val);
  }
}

// ---------------- fused per-tile SAGE-LSTM aggregation (16 steps), weights LDS-resident via TDM ----------------
__global__ __launch_bounds__(256) void k_lstm_sage(
    const unsigned short* __restrict__ h2b, const long long* __restrict__ src,
    const unsigned short* __restrict__ wih, const unsigned short* __restrict__ whh,
    const float* __restrict__ bih, const float* __restrict__ bhh,
    unsigned short* __restrict__ houtb) {
  extern __shared__ unsigned char smem[];
  unsigned short* Wih_s = (unsigned short*)smem;           // 512*136
  unsigned short* Whh_s = Wih_s + 512 * 136;               // 512*136
  unsigned short* xt    = Whh_s + 512 * 136;               // 16*136
  unsigned short* hbt   = xt + 16 * 136;                   // 16*136
  float* hf = (float*)(hbt + 16 * 136);                    // 16*132
  float* cf = hf + 16 * 132;                               // 16*132

  int node0 = blockIdx.x * 16;
  int tid = threadIdx.x;
#if HAVE_TDM
  if (tid < 32) {   // wave 0 drives the Tensor Data Mover for both weight tiles
    tdm_load_2d_pad(wih, (unsigned int)(size_t)(void*)Wih_s, 512, CD);
    tdm_load_2d_pad(whh, (unsigned int)(size_t)(void*)Whh_s, 512, CD);
    __builtin_amdgcn_s_wait_tensorcnt(0);
  }
#else
  for (int idx = tid; idx < 512 * CD; idx += 256) {
    int r = idx >> 7, c = idx & 127;
    Wih_s[r * 136 + c] = wih[idx];
    Whh_s[r * 136 + c] = whh[idx];
  }
#endif
  for (int idx = tid; idx < 16 * 132; idx += 256) { hf[idx] = 0.f; cf[idx] = 0.f; }
  __syncthreads();

  int wv = tid >> 5, lane = tid & 31;
  int jb = wv * 16;
  int j = jb + (lane & 15);
  int mo = (lane & 16) ? 8 : 0;
  float bi  = bih[0 * CD + j] + bhh[0 * CD + j];
  float bff = bih[1 * CD + j] + bhh[1 * CD + j];
  float bg  = bih[2 * CD + j] + bhh[2 * CD + j];
  float bo  = bih[3 * CD + j] + bhh[3 * CD + j];

  const int grow = tid >> 4, gcb = (tid & 15) * 8;   // gather row/col for this thread
  for (int t = 0; t < KDEG; ++t) {
    {   // gather x_t tile (b128) and refresh bf16 h tile
      long long s = src[(size_t)(node0 + grow) * KDEG + t];
      if (t + 1 < KDEG) __builtin_prefetch(h2b + (size_t)src[(size_t)(node0 + grow) * KDEG + t + 1] * CD + gcb, 0, 1);
      *(uint4*)(xt + grow * 136 + gcb) = *(const uint4*)(h2b + (size_t)s * CD + gcb);
      U8 u;
#pragma unroll
      for (int e = 0; e < 8; ++e) u.s[e] = f2bf(hf[grow * 132 + gcb + e]);
      *(uint4*)(hbt + grow * 136 + gcb) = u.v;
    }
    __syncthreads();
    v8f zi = {}, zf = {}, zg = {}, zo = {};
#pragma unroll
    for (int kb = 0; kb < CD; kb += 32) {
      v16bf ax = load_frag(xt, 136, 0, kb, lane);
      v16bf ah = load_frag(hbt, 136, 0, kb, lane);
      zi = wmma_bf16(ax, load_frag(Wih_s, 136, 0 * CD + jb, kb, lane), zi);
      zi = wmma_bf16(ah, load_frag(Whh_s, 136, 0 * CD + jb, kb, lane), zi);
      zf = wmma_bf16(ax, load_frag(Wih_s, 136, 1 * CD + jb, kb, lane), zf);
      zf = wmma_bf16(ah, load_frag(Whh_s, 136, 1 * CD + jb, kb, lane), zf);
      zg = wmma_bf16(ax, load_frag(Wih_s, 136, 2 * CD + jb, kb, lane), zg);
      zg = wmma_bf16(ah, load_frag(Whh_s, 136, 2 * CD + jb, kb, lane), zg);
      zo = wmma_bf16(ax, load_frag(Wih_s, 136, 3 * CD + jb, kb, lane), zo);
      zo = wmma_bf16(ah, load_frag(Whh_s, 136, 3 * CD + jb, kb, lane), zo);
    }
#pragma unroll
    for (int v = 0; v < 8; ++v) {
      int m = v + mo;
      float iv = sigm(zi[v] + bi);
      float fv = sigm(zf[v] + bff);
      float gv = tanhf(zg[v] + bg);
      float ov = sigm(zo[v] + bo);
      float c = fv * cf[m * 132 + j] + iv * gv;
      cf[m * 132 + j] = c;
      hf[m * 132 + j] = ov * tanhf(c);
    }
    __syncthreads();
  }
  {
    U8 o;
#pragma unroll
    for (int e = 0; e < 8; ++e) o.s[e] = f2bf(hf[grow * 132 + gcb + e]);
    *(uint4*)(houtb + (size_t)(node0 + grow) * CD + gcb) = o.v;
  }
}

// ---------------- fused per-tile JK bi-LSTM + attention ----------------
__global__ __launch_bounds__(192) void k_jk(
    const unsigned short* __restrict__ h1b, const unsigned short* __restrict__ h2b,
    const unsigned short* __restrict__ h3b,
    const unsigned short* __restrict__ wihf, const unsigned short* __restrict__ whhf,
    const float* __restrict__ bihf, const float* __restrict__ bhhf,
    const unsigned short* __restrict__ wihbk, const unsigned short* __restrict__ whhbk,
    const float* __restrict__ bihb, const float* __restrict__ bhhb,
    const float* __restrict__ attw, const float* __restrict__ attb,
    float* __restrict__ out) {
  extern __shared__ unsigned char smem[];
  unsigned short* xs0  = (unsigned short*)smem;      // 16*136 each
  unsigned short* xs1  = xs0 + 16 * 136;
  unsigned short* xs2  = xs1 + 16 * 136;
  unsigned short* hcur = xs2 + 16 * 136;             // 16*200
  float* hf = (float*)(hcur + 16 * 200);             // 16*196
  float* cf = hf + 16 * 196;                         // 16*196
  float* sc = cf + 16 * 196;                         // 16*3

  int node0 = blockIdx.x * 16;
  int tid = threadIdx.x;
  const unsigned short* hsrc[3] = {h1b, h2b, h3b};
  unsigned short* xsl[3] = {xs0, xs1, xs2};
  for (int l = 0; l < 3; ++l)
    for (int idx = tid; idx < 16 * 16; idx += 192) {
      int r = idx >> 4, cb = (idx & 15) * 8;
      *(uint4*)(xsl[l] + r * 136 + cb) = *(const uint4*)(hsrc[l] + (size_t)(node0 + r) * CD + cb);
    }
  for (int idx = tid; idx < 48; idx += 192) sc[idx] = attb[0];
  __syncthreads();

  int wv = tid >> 5, lane = tid & 31;
  int mo = (lane & 16) ? 8 : 0;

  for (int d = 0; d < 2; ++d) {
    const unsigned short* Wih = d ? wihbk : wihf;
    const unsigned short* Whh = d ? whhbk : whhf;
    const float* bih = d ? bihb : bihf;
    const float* bhh = d ? bhhb : bhhf;
    for (int idx = tid; idx < 16 * 196; idx += 192) { hf[idx] = 0.f; cf[idx] = 0.f; }
    __syncthreads();
    for (int tt = 0; tt < 3; ++tt) {
      int xi = d ? (2 - tt) : tt;                // sequence index consumed; also alpha slot
      const unsigned short* xsrc = xsl[xi];
      for (int idx = tid; idx < 16 * JKH; idx += 192) {
        int r = idx / JKH, c = idx % JKH;
        hcur[r * 200 + c] = f2bf(hf[r * 196 + c]);
      }
      __syncthreads();
#pragma unroll
      for (int jt = 0; jt < 2; ++jt) {
        int jb = (wv * 2 + jt) * 16;
        int j = jb + (lane & 15);
        v8f zi = {}, zf2 = {}, zg = {}, zo = {};
#pragma unroll
        for (int kb = 0; kb < CD; kb += 32) {
          v16bf ax = load_frag(xsrc, 136, 0, kb, lane);
          zi  = wmma_bf16(ax, load_frag(Wih, CD, 0 * JKH + jb, kb, lane), zi);
          zf2 = wmma_bf16(ax, load_frag(Wih, CD, 1 * JKH + jb, kb, lane), zf2);
          zg  = wmma_bf16(ax, load_frag(Wih, CD, 2 * JKH + jb, kb, lane), zg);
          zo  = wmma_bf16(ax, load_frag(Wih, CD, 3 * JKH + jb, kb, lane), zo);
        }
#pragma unroll
        for (int kb = 0; kb < JKH; kb += 32) {
          v16bf ah = load_frag(hcur, 200, 0, kb, lane);
          zi  = wmma_bf16(ah, load_frag(Whh, JKH, 0 * JKH + jb, kb, lane), zi);
          zf2 = wmma_bf16(ah, load_frag(Whh, JKH, 1 * JKH + jb, kb, lane), zf2);
          zg  = wmma_bf16(ah, load_frag(Whh, JKH, 2 * JKH + jb, kb, lane), zg);
          zo  = wmma_bf16(ah, load_frag(Whh, JKH, 3 * JKH + jb, kb, lane), zo);
        }
        float bi  = bih[0 * JKH + j] + bhh[0 * JKH + j];
        float bff = bih[1 * JKH + j] + bhh[1 * JKH + j];
        float bg  = bih[2 * JKH + j] + bhh[2 * JKH + j];
        float bo  = bih[3 * JKH + j] + bhh[3 * JKH + j];
        float aw  = attw[d * JKH + j];
#pragma unroll
        for (int v = 0; v < 8; ++v) {
          int m = v + mo;
          float iv = sigm(zi[v] + bi);
          float fv = sigm(zf2[v] + bff);
          float gv = tanhf(zg[v] + bg);
          float ov = sigm(zo[v] + bo);
          float c = fv * cf[m * 196 + j] + iv * gv;
          cf[m * 196 + j] = c;
          float h = ov * tanhf(c);
          hf[m * 196 + j] = h;
          atomicAdd(&sc[m * 3 + xi], aw * h);    // incremental attention score
        }
      }
      __syncthreads();
    }
    __syncthreads();
  }
  if (tid < 16) {
    float s0 = sc[tid * 3 + 0], s1 = sc[tid * 3 + 1], s2 = sc[tid * 3 + 2];
    float mx = fmaxf(s0, fmaxf(s1, s2));
    float e0 = expf(s0 - mx), e1 = expf(s1 - mx), e2 = expf(s2 - mx);
    float inv = 1.0f / (e0 + e1 + e2);
    sc[tid * 3 + 0] = e0 * inv; sc[tid * 3 + 1] = e1 * inv; sc[tid * 3 + 2] = e2 * inv;
  }
  __syncthreads();
  for (int idx = tid; idx < 16 * CD; idx += 192) {
    int m = idx >> 7, f = idx & 127;
    float val = sc[m * 3 + 0] * bf2f(xs0[m * 136 + f])
              + sc[m * 3 + 1] * bf2f(xs1[m * 136 + f])
              + sc[m * 3 + 2] * bf2f(xs2[m * 136 + f]);
    out[(size_t)(node0 + m) * CD + f] = val;
  }
}

// ---------------- host ----------------
extern "C" void kernel_launch(void* const* d_in, const int* in_sizes, int n_in,
                              void* d_out, int out_size, void* d_ws, size_t ws_size,
                              hipStream_t stream) {
  (void)in_sizes; (void)n_in; (void)out_size; (void)ws_size;
  const float* x      = (const float*)d_in[0];
  const long long* ei = (const long long*)d_in[1];   // int64 [2,E]; src = first E
  const float* Wl1 = (const float*)d_in[2];
  const float* bl1 = (const float*)d_in[3];
  const float* Wr1 = (const float*)d_in[4];
  const float* Wl2 = (const float*)d_in[5];
  const float* bl2 = (const float*)d_in[6];
  const float* Wr2 = (const float*)d_in[7];
  const float* Wl3 = (const float*)d_in[8];
  const float* bl3 = (const float*)d_in[9];
  const float* Wr3 = (const float*)d_in[10];
  const float* Wih_a = (const float*)d_in[11];
  const float* Whh_a = (const float*)d_in[12];
  const float* bih_a = (const float*)d_in[13];
  const float* bhh_a = (const float*)d_in[14];
  const float* Wih_f = (const float*)d_in[15];
  const float* Whh_f = (const float*)d_in[16];
  const float* bih_f = (const float*)d_in[17];
  const float* bhh_f = (const float*)d_in[18];
  const float* Wih_b = (const float*)d_in[19];
  const float* Whh_b = (const float*)d_in[20];
  const float* bih_b = (const float*)d_in[21];
  const float* bhh_b = (const float*)d_in[22];
  const float* att_w = (const float*)d_in[23];
  const float* att_b = (const float*)d_in[24];
  const long long* src = ei;

  unsigned short* ws = (unsigned short*)d_ws;
  size_t o = 0;
  auto take = [&](size_t n) { unsigned short* p = ws + o; o += n; return p; };
  unsigned short* xb   = take((size_t)NN * CD);
  unsigned short* h1b  = take((size_t)NN * CD);
  unsigned short* h2b  = take((size_t)NN * CD);
  unsigned short* h3b  = take((size_t)NN * CD);
  unsigned short* aggb = take((size_t)NN * CD);
  unsigned short* lhb  = take((size_t)NN * CD);
  unsigned short* wl1b = take(CD * CD);
  unsigned short* wr1b = take(CD * CD);
  unsigned short* wl2b = take(CD * CD);
  unsigned short* wr2b = take(CD * CD);
  unsigned short* wl3b = take(CD * CD);
  unsigned short* wr3b = take(CD * CD);
  unsigned short* wihab = take(4 * CD * CD);
  unsigned short* whhab = take(4 * CD * CD);
  unsigned short* wihfb = take(4 * JKH * CD);
  unsigned short* whhfb = take(4 * JKH * JKH);
  unsigned short* wihbb = take(4 * JKH * CD);
  unsigned short* whhbb = take(4 * JKH * JKH);

  auto cvt = [&](const float* s, unsigned short* dptr, int n) {
    k_cvt<<<(n + 255) / 256, 256, 0, stream>>>(s, dptr, n);
  };
  cvt(x, xb, NN * CD);
  cvt(Wl1, wl1b, CD * CD);  cvt(Wr1, wr1b, CD * CD);
  cvt(Wl2, wl2b, CD * CD);  cvt(Wr2, wr2b, CD * CD);
  cvt(Wl3, wl3b, CD * CD);  cvt(Wr3, wr3b, CD * CD);
  cvt(Wih_a, wihab, 4 * CD * CD);  cvt(Whh_a, whhab, 4 * CD * CD);
  cvt(Wih_f, wihfb, 4 * JKH * CD); cvt(Whh_f, whhfb, 4 * JKH * JKH);
  cvt(Wih_b, wihbb, 4 * JKH * CD); cvt(Whh_b, whhbb, 4 * JKH * JKH);

  // layer 1: mean aggregation + combine + relu
  k_mean_agg<<<NT, 256, 0, stream>>>(xb, src, aggb);
  k_sage_gemm<<<NT, 256, 0, stream>>>(aggb, wl1b, bl1, xb, wr1b, h1b, 1);
  // layer 2: max aggregation + combine + relu
  k_max_agg<<<NT, 256, 0, stream>>>(h1b, src, aggb);
  k_sage_gemm<<<NT, 256, 0, stream>>>(aggb, wl2b, bl2, h1b, wr2b, h2b, 1);
  // layer 3: fused LSTM aggregation (TDM-staged LDS weights, 16 steps), then combine
  size_t lstm_lds = (size_t)(512 * 136 * 2 + 16 * 136 * 2) * 2 + (size_t)(16 * 132 * 2) * 4;
  k_lstm_sage<<<NT, 256, lstm_lds, stream>>>(h2b, src, wihab, whhab, bih_a, bhh_a, lhb);
  k_sage_gemm<<<NT, 256, 0, stream>>>(lhb, wl3b, bl3, h2b, wr3b, h3b, 0);
  // JumpingKnowledge bi-LSTM + attention -> f32 output
  size_t jk_lds = (size_t)(3 * 16 * 136 + 16 * 200) * 2 + (size_t)(16 * 196 * 2 + 48) * 4;
  k_jk<<<NT, 192, jk_lds, stream>>>(h1b, h2b, h3b, wihfb, whhfb, bih_f, bhh_f,
                                    wihbb, whhbb, bih_b, bhh_b, att_w, att_b,
                                    (float*)d_out);
}
